// GraphTransformerLayer_45930380264133
// MI455X (gfx1250) — matrix operand order repaired and checked by reading, not compiled, not run on hardware
//
#include <hip/hip_runtime.h>

#define NN 50000
#define EE 800000
#define DD 128
#define HH 8
#define CC 16
#define FF 512

typedef __bf16 bf16;
typedef __attribute__((ext_vector_type(16))) __bf16 bf16x16;
typedef __attribute__((ext_vector_type(8)))  __bf16 bf16x8;
typedef __attribute__((ext_vector_type(8)))  float  floatx8;

__device__ inline floatx8 wmma_bf16(bf16x16 a, bf16x16 b, floatx8 c) {
  // (neg_a, A, neg_b, B, c_mod, C, reuse_a, reuse_b)
  return __builtin_amdgcn_wmma_f32_16x16x32_bf16(false, a, false, b, (short)0, c, false, false);
}

__device__ inline bf16x16 load_a_frag(const bf16* rowptr, int ka) {
  // A 16x32 bf16 layout: elems 0..15 -> K=ka..ka+15 (contiguous)
  bf16x8 lo = *(const bf16x8*)(rowptr + ka);
  bf16x8 hi = *(const bf16x8*)(rowptr + ka + 8);
  return __builtin_shufflevector(lo, hi, 0,1,2,3,4,5,6,7,8,9,10,11,12,13,14,15);
}

__device__ inline void atomic_max_f32(float* addr, float v) {
  if (v >= 0.0f) atomicMax((int*)addr, __float_as_int(v));
  else           atomicMin((unsigned int*)addr, __float_as_uint(v));
}

// ---------------------------------------------------------------------------
// 1) Convert weights to bf16, transposed to [n][k] so B fragments are
//    contiguous 32-byte per-lane loads.
//    ws_w halfs: [0,65536) Wq/Wk/Wv/Wskip^T (4x128x128), [65536,131072) W1^T
//    (512x128), [131072,196608) W2^T (128x512).
// ---------------------------------------------------------------------------
__global__ __launch_bounds__(256) void prep_weights(
    const float* __restrict__ Wq, const float* __restrict__ Wk,
    const float* __restrict__ Wv, const float* __restrict__ Ws,
    const float* __restrict__ W1, const float* __restrict__ W2,
    unsigned short* __restrict__ ws_w_u) {
  bf16* ws_w = (bf16*)ws_w_u;
  int t = blockIdx.x * 256 + threadIdx.x;
  if (t < 65536) {
    int mat = t >> 14, i = t & 16383;
    int n = i >> 7, k = i & 127;
    const float* W = (mat == 0) ? Wq : (mat == 1) ? Wk : (mat == 2) ? Wv : Ws;
    ws_w[t] = (bf16)W[k * DD + n];
  } else if (t < 131072) {
    int i = t - 65536; int n = i >> 7, k = i & 127;     // n<512, k<128
    ws_w[t] = (bf16)W1[k * FF + n];
  } else if (t < 196608) {
    int i = t - 131072; int n = i >> 9, k = i & 511;    // n<128, k<512
    ws_w[t] = (bf16)W2[k * DD + n];
  }
}

// ---------------------------------------------------------------------------
// 2) Init accumulation buffers every call (harness poisons ws once).
// ---------------------------------------------------------------------------
__global__ __launch_bounds__(256) void init_bufs(
    float* __restrict__ outagg, float* __restrict__ amax,
    float* __restrict__ denom, float* __restrict__ stats) {
  int t = blockIdx.x * 256 + threadIdx.x;
  if (t < NN * DD) outagg[t] = 0.0f;
  if (t < NN * HH) { amax[t] = -__builtin_inff(); denom[t] = 0.0f; }
  if (t < 256) stats[t] = 0.0f;
}

// ---------------------------------------------------------------------------
// 3) Q/K/V/skip GEMM: one 16-row tile per block; wave w does matrix w>>1,
//    column tiles (w&1)*4 .. +3. 16 WMMAs per wave.
// ---------------------------------------------------------------------------
__global__ __launch_bounds__(256) void qkv_gemm(
    const float* __restrict__ x, const unsigned short* __restrict__ Wt_u,
    const float* __restrict__ bq, const float* __restrict__ bk,
    const float* __restrict__ bv, const float* __restrict__ bs,
    float* __restrict__ qb, float* __restrict__ kb,
    float* __restrict__ vb, float* __restrict__ xrb) {
  __shared__ bf16 xa[16 * DD];
  const bf16* Wt = (const bf16*)Wt_u;
  int rowbase = blockIdx.x * 16;
  int tid = threadIdx.x;
  {
    int r = tid >> 4, c0 = (tid & 15) * 8;
    const float* src = x + (size_t)(rowbase + r) * DD + c0;
#pragma unroll
    for (int i = 0; i < 8; i++) xa[r * DD + c0 + i] = (bf16)src[i];
  }
  __syncthreads();
  int wave = tid >> 5, lane = tid & 31;
  int mat = wave >> 1, tile0 = (wave & 1) * 4;
  const bf16* Wmt = Wt + (size_t)mat * DD * DD;   // [n][k]
  int m = lane & 15;
  int ka  = (lane >> 4) * 16;
  int kbh = (lane >> 4) * 16;
  floatx8 acc[4] = {};
#pragma unroll
  for (int ks = 0; ks < 4; ks++) {
    int kb0 = ks * 32;
    bf16x16 afrag = load_a_frag(&xa[m * DD + kb0], ka);
#pragma unroll
    for (int tt = 0; tt < 4; tt++) {
      int n = (tile0 + tt) * 16 + m;
      bf16x16 bfrag = *(const bf16x16*)(Wmt + (size_t)n * DD + kb0 + kbh);
      acc[tt] = wmma_bf16(afrag, bfrag, acc[tt]);
    }
  }
  float* outp = (mat == 0) ? qb : (mat == 1) ? kb : (mat == 2) ? vb : xrb;
  const float* bias = (mat == 0) ? bq : (mat == 1) ? bk : (mat == 2) ? bv : bs;
  int rsub = (lane >> 4) * 8;
#pragma unroll
  for (int tt = 0; tt < 4; tt++) {
    int col = (tile0 + tt) * 16 + m;
    float bval = bias[col];
#pragma unroll
    for (int r = 0; r < 8; r++)
      outp[(size_t)(rowbase + rsub + r) * DD + col] = acc[tt][r] + bval;
  }
}

// ---------------------------------------------------------------------------
// 4) Edge pass 1: alpha = <q[dst], k[src]+e> / 4 ; segment max via atomics.
//    One thread per (edge, head); e recomputed from K=10 projection (We in L0).
// ---------------------------------------------------------------------------
__global__ __launch_bounds__(256) void edge_alpha(
    const int* __restrict__ eidx, const float* __restrict__ eattr,
    const float* __restrict__ We, const float* __restrict__ qb,
    const float* __restrict__ kb, float* __restrict__ alphab,
    float* __restrict__ amax) {
  int t = blockIdx.x * 256 + threadIdx.x;
  if (t >= EE * HH) return;
  int e = t >> 3, h = t & 7;
  int s = eidx[e], d = eidx[EE + e];
  float ea[10];
#pragma unroll
  for (int j = 0; j < 10; j++) ea[j] = eattr[e * 10 + j];
  const float* qrow = qb + (size_t)d * DD + h * CC;
  const float* krow = kb + (size_t)s * DD + h * CC;
  const float* Wec  = We + h * CC;
  float acc = 0.0f;
#pragma unroll
  for (int c = 0; c < CC; c++) {
    float ec = 0.0f;
#pragma unroll
    for (int j = 0; j < 10; j++) ec += ea[j] * Wec[j * DD + c];
    acc += qrow[c] * (krow[c] + ec);
  }
  acc *= 0.25f;   // 1/sqrt(16)
  alphab[t] = acc;
  atomic_max_f32(&amax[d * HH + h], acc);
}

// ---------------------------------------------------------------------------
// 5) Edge pass 2: ex = exp(alpha - amax[dst]); segment sum.
// ---------------------------------------------------------------------------
__global__ __launch_bounds__(256) void edge_expsum(
    const int* __restrict__ eidx, float* __restrict__ alphab,
    const float* __restrict__ amax, float* __restrict__ denom) {
  int t = blockIdx.x * 256 + threadIdx.x;
  if (t >= EE * HH) return;
  int e = t >> 3, h = t & 7;
  int d = eidx[EE + e];
  float ex = expf(alphab[t] - amax[d * HH + h]);
  alphab[t] = ex;
  atomicAdd(&denom[d * HH + h], ex);
}

// ---------------------------------------------------------------------------
// 6) Edge pass 3: scatter (v[src]+e) * alpha_norm into out[dst].
// ---------------------------------------------------------------------------
__global__ __launch_bounds__(256) void edge_message(
    const int* __restrict__ eidx, const float* __restrict__ eattr,
    const float* __restrict__ We, const float* __restrict__ vb,
    const float* __restrict__ alphab, const float* __restrict__ denom,
    float* __restrict__ outagg) {
  int t = blockIdx.x * 256 + threadIdx.x;
  if (t >= EE * HH) return;
  int e = t >> 3, h = t & 7;
  int s = eidx[e], d = eidx[EE + e];
  float w = alphab[t] / (denom[d * HH + h] + 1e-16f);
  float ea[10];
#pragma unroll
  for (int j = 0; j < 10; j++) ea[j] = eattr[e * 10 + j];
  const float* vrow = vb + (size_t)s * DD + h * CC;
  const float* Wec  = We + h * CC;
  float* orow = outagg + (size_t)d * DD + h * CC;
#pragma unroll
  for (int c = 0; c < CC; c++) {
    float ec = 0.0f;
#pragma unroll
    for (int j = 0; j < 10; j++) ec += ea[j] * Wec[j * DD + c];
    atomicAdd(&orow[c], (vrow[c] + ec) * w);
  }
}

// ---------------------------------------------------------------------------
// 7) Beta gate + residual; accumulate per-feature sum(h), sum(h^2).
//    128 threads = one feature each; wave shfl + LDS reduction for the
//    384-wide beta dot product.
// ---------------------------------------------------------------------------
__global__ __launch_bounds__(128) void beta_combine(
    const float* __restrict__ x, const float* __restrict__ outagg,
    const float* __restrict__ xr, const float* __restrict__ Wb,
    float* __restrict__ hbuf, float* __restrict__ colsum,
    float* __restrict__ colsq) {
  __shared__ float red[4];
  int f = threadIdx.x;
  int lane = f & 31, wv = f >> 5;
  float w0 = Wb[f], w1 = Wb[DD + f], w2 = Wb[2 * DD + f];
  float ls = 0.0f, lq = 0.0f;
  for (int n = blockIdx.x; n < NN; n += gridDim.x) {
    size_t base = (size_t)n * DD + f;
    float o = outagg[base], xrv = xr[base];
    float p = o * w0 + xrv * w1 + (o - xrv) * w2;
#pragma unroll
    for (int off = 16; off; off >>= 1) p += __shfl_xor(p, off, 32);
    if (lane == 0) red[wv] = p;
    __syncthreads();
    float total = red[0] + red[1] + red[2] + red[3];
    __syncthreads();
    float beta = 1.0f / (1.0f + expf(-total));
    float hval = x[base] + beta * xrv + (1.0f - beta) * o;
    hbuf[base] = hval;
    ls += hval; lq += hval * hval;
  }
  atomicAdd(&colsum[f], ls);
  atomicAdd(&colsq[f], lq);
}

// ---------------------------------------------------------------------------
// 8) GraphNorm stats: fold normalize into per-feature affine (a, b).
// ---------------------------------------------------------------------------
__global__ void finalize_stats(
    const float* __restrict__ colsum, const float* __restrict__ colsq,
    const float* __restrict__ gn_w, const float* __restrict__ gn_b,
    const float* __restrict__ gn_ms, float* __restrict__ af,
    float* __restrict__ bfv) {
  int f = threadIdx.x;
  float invN = 1.0f / (float)NN;
  float mean = colsum[f] * invN;
  float ms = gn_ms[f];
  float ex2 = colsq[f] * invN;
  float var = ex2 - 2.0f * ms * mean * mean + ms * ms * mean * mean;
  float a = gn_w[f] * rsqrtf(var + 1e-5f);
  af[f] = a;
  bfv[f] = gn_b[f] - ms * mean * a;
}

// ---------------------------------------------------------------------------
// 9) Fused norm + FFN: hn = a*h+b ; GEMM1 (128->512) + bias + exact GELU
//    into LDS bf16 ; GEMM2 (512->128) + bias + residual -> d_out.
// ---------------------------------------------------------------------------
__global__ __launch_bounds__(256) void ffn_kernel(
    const float* __restrict__ hbuf, const float* __restrict__ af,
    const float* __restrict__ bfv, const unsigned short* __restrict__ W1t_u,
    const float* __restrict__ b1, const unsigned short* __restrict__ W2t_u,
    const float* __restrict__ b2, float* __restrict__ out) {
  __shared__ bf16  hn_bf[16 * DD];
  __shared__ float hn_f [16 * DD];
  __shared__ bf16  g    [16 * FF];
  const bf16* W1t = (const bf16*)W1t_u;   // [512][128]
  const bf16* W2t = (const bf16*)W2t_u;   // [128][512]
  int rowbase = blockIdx.x * 16;
  int tid = threadIdx.x;
  {
    int r = tid >> 4, c0 = (tid & 15) * 8;
    const float* src = hbuf + (size_t)(rowbase + r) * DD + c0;
#pragma unroll
    for (int i = 0; i < 8; i++) {
      float v = src[i] * af[c0 + i] + bfv[c0 + i];
      hn_f [r * DD + c0 + i] = v;
      hn_bf[r * DD + c0 + i] = (bf16)v;
    }
  }
  __syncthreads();
  int wave = tid >> 5, lane = tid & 31;
  int m = lane & 15;
  int ka = (lane >> 4) * 16, kbh = (lane >> 4) * 16, rsub = (lane >> 4) * 8;
  // GEMM1: 16x512, wave handles 4 of 32 column tiles.
  floatx8 acc[4] = {};
#pragma unroll
  for (int ks = 0; ks < 4; ks++) {
    int kb0 = ks * 32;
    bf16x16 afrag = load_a_frag(&hn_bf[m * DD + kb0], ka);
#pragma unroll
    for (int tt = 0; tt < 4; tt++) {
      int n = (wave * 4 + tt) * 16 + m;
      bf16x16 bfrag = *(const bf16x16*)(W1t + (size_t)n * DD + kb0 + kbh);
      acc[tt] = wmma_bf16(afrag, bfrag, acc[tt]);
    }
  }
#pragma unroll
  for (int tt = 0; tt < 4; tt++) {
    int col = (wave * 4 + tt) * 16 + m;
    float bb = b1[col];
#pragma unroll
    for (int r = 0; r < 8; r++) {
      float v = acc[tt][r] + bb;
      v = 0.5f * v * (1.0f + erff(v * 0.70710678118654752f));   // exact GELU
      g[(rsub + r) * FF + col] = (bf16)v;
    }
  }
  __syncthreads();
  // GEMM2: 16x128, wave handles one column tile, K=512 (16 steps).
  floatx8 acc2 = {};
#pragma unroll
  for (int ks = 0; ks < 16; ks++) {
    int kb0 = ks * 32;
    bf16x16 afrag = load_a_frag(&g[m * FF + kb0], ka);
    int n = wave * 16 + m;
    bf16x16 bfrag = *(const bf16x16*)(W2t + (size_t)n * FF + kb0 + kbh);
    acc2 = wmma_bf16(afrag, bfrag, acc2);
  }
  {
    int col = wave * 16 + m;
    float bb = b2[col];
#pragma unroll
    for (int r = 0; r < 8; r++) {
      int lr = rsub + r;
      out[(size_t)(rowbase + lr) * DD + col] = acc2[r] + bb + hn_f[lr * DD + col];
    }
  }
}

// ---------------------------------------------------------------------------
extern "C" void kernel_launch(void* const* d_in, const int* in_sizes, int n_in,
                              void* d_out, int out_size, void* d_ws, size_t ws_size,
                              hipStream_t stream) {
  const float* x      = (const float*)d_in[0];
  const int*   eidx   = (const int*)  d_in[1];
  const float* eattr  = (const float*)d_in[2];
  const float* Wq     = (const float*)d_in[3];
  const float* bq     = (const float*)d_in[4];
  const float* Wk     = (const float*)d_in[5];
  const float* bk     = (const float*)d_in[6];
  const float* Wv     = (const float*)d_in[7];
  const float* bv     = (const float*)d_in[8];
  const float* We     = (const float*)d_in[9];
  const float* Wskip  = (const float*)d_in[10];
  const float* bskip  = (const float*)d_in[11];
  const float* Wbeta  = (const float*)d_in[12];
  const float* gn_w   = (const float*)d_in[13];
  const float* gn_b   = (const float*)d_in[14];
  const float* gn_ms  = (const float*)d_in[15];
  const float* W1     = (const float*)d_in[16];
  const float* b1     = (const float*)d_in[17];
  const float* W2     = (const float*)d_in[18];
  const float* b2     = (const float*)d_in[19];
  float* outp = (float*)d_out;

  char* ws = (char*)d_ws;
  const size_t NB = (size_t)NN * DD * sizeof(float);      // 25.6 MB
  unsigned short* ws_w = (unsigned short*)ws;             // 393216 B of bf16 weights
  size_t off = 393216;
  float* qb     = (float*)(ws + off); off += NB;
  float* kb     = (float*)(ws + off); off += NB;
  float* vb     = (float*)(ws + off); off += NB;
  float* xrb    = (float*)(ws + off); off += NB;
  float* outagg = (float*)(ws + off); off += NB;
  float* hbuf   = (float*)(ws + off); off += NB;
  float* alphab = (float*)(ws + off); off += (size_t)EE * HH * sizeof(float);
  float* amax   = (float*)(ws + off); off += (size_t)NN * HH * sizeof(float);
  float* denom  = (float*)(ws + off); off += (size_t)NN * HH * sizeof(float);
  float* colsum = (float*)(ws + off); off += 512;
  float* colsq  = (float*)(ws + off); off += 512;
  float* afv    = (float*)(ws + off); off += 512;
  float* bfv    = (float*)(ws + off); off += 512;

  unsigned short* Wt4 = ws_w;                 // [4][128][128] transposed
  unsigned short* W1t = ws_w + 65536;         // [512][128]
  unsigned short* W2t = ws_w + 131072;        // [128][512]

  prep_weights<<<768, 256, 0, stream>>>(Wq, Wk, Wv, Wskip, W1, W2, ws_w);
  init_bufs<<<(NN * DD + 255) / 256, 256, 0, stream>>>(outagg, amax, denom, colsum);
  qkv_gemm<<<NN / 16, 256, 0, stream>>>(x, Wt4, bq, bk, bv, bskip, qb, kb, vb, xrb);

  int egrid = (EE * HH + 255) / 256;
  edge_alpha  <<<egrid, 256, 0, stream>>>(eidx, eattr, We, qb, kb, alphab, amax);
  edge_expsum <<<egrid, 256, 0, stream>>>(eidx, alphab, amax, denom);
  edge_message<<<egrid, 256, 0, stream>>>(eidx, eattr, We, vb, alphab, denom, outagg);

  beta_combine<<<512, 128, 0, stream>>>(x, outagg, xrb, Wbeta, hbuf, colsum, colsq);
  finalize_stats<<<1, 128, 0, stream>>>(colsum, colsq, gn_w, gn_b, gn_ms, afv, bfv);
  ffn_kernel<<<NN / 16, 256, 0, stream>>>(hbuf, afv, bfv, W1t, b1, W2t, b2, outp);

  (void)in_sizes; (void)n_in; (void)out_size; (void)ws_size;
}
